// EncodeProcessDecode_59459527246187
// MI455X (gfx1250) — compile-verified, matrix operand order checked by reading, not run on hardware
//
#include <hip/hip_runtime.h>
#include <hip/hip_bf16.h>

// ---------------------------------------------------------------------------
// CDNA5 (gfx1250) fused GraphNet Encode-Process-Decode using bf16 WMMA.
//   v_wmma_f32_16x16x32_bf16 : D(16x16 f32) = A(16x32 bf16) x B(32x16 bf16) + C
// One block = 256 threads = 8 wave32. Block processes 32 rows; each wave owns
// one 16-column tile and computes TWO 16x16 output tiles (rows 0-15 / 16-31)
// off a single B fragment -> 2x weight reuse + two independent WMMA chains.
// Activations that feed WMMA (h, e) are mirrored in bf16 so edge gathers move
// half the bytes; the f32 master copy of h is kept for the residual path.
// ---------------------------------------------------------------------------

typedef __attribute__((ext_vector_type(16))) __bf16        bf16x16;
typedef __attribute__((ext_vector_type(8)))  float         v8f;
typedef __attribute__((ext_vector_type(8)))  unsigned int  v8u;

// K-index pattern within a 16x32 bf16 A/B fragment (per ISA 7.12.2):
//  lanes 0-15  : VGPR0..3 hold K=0..7 (2/VGPR), VGPR4..7 hold K=16..23
//  lanes 16-31 : VGPR0..3 hold K=8..15,        VGPR4..7 hold K=24..31
__device__ __forceinline__ int frag_kbase(int half, int v) {
    return (v < 4) ? (half * 8 + v * 2) : (16 + half * 8 + (v - 4) * 2);
}

// ---------------------------------------------------------------------------
// Weight pre-shuffle: f32 row-major W[K][Nc] -> bf16 fragment-ordered buffer
// dst[((kc*colTiles + ct)*32 + lane)*16 + i], zero-padded to Kpad/Npad.
// ---------------------------------------------------------------------------
__global__ void k_convert_w(const float* __restrict__ W, __bf16* __restrict__ dst,
                            int K, int Kpad, int Nc, int Npad) {
    int idx = blockIdx.x * 256 + threadIdx.x;
    int colTiles = Npad >> 4;
    int total = (Kpad >> 5) * colTiles * 512;
    if (idx >= total) return;
    int i    = idx & 15;
    int lane = (idx >> 4) & 31;
    int frag = idx >> 9;
    int kc   = frag / colTiles;
    int ct   = frag - kc * colTiles;
    int half = lane >> 4;
    int v    = i >> 1;
    int w    = i & 1;
    int k    = kc * 32 + frag_kbase(half, v) + w;
    int col  = ct * 16 + (lane & 15);
    float val = (k < K && col < Nc) ? W[(size_t)k * Nc + col] : 0.0f;
    dst[idx] = (__bf16)val;
}

__global__ void k_zero(float* __restrict__ p, long n) {
    long i = (long)blockIdx.x * 256 + threadIdx.x;
    if (i < n) p[i] = 0.0f;
}

// ---------------------------------------------------------------------------
// Wave-level GEMM: two 16x16 tiles (row-halves) sharing one B fragment.
//   A: LDS bf16 [32 x K] row-major (stride aStride), B: fragment-shuffled.
// ---------------------------------------------------------------------------
__device__ __forceinline__ void mm_tile2(const __bf16* a, int aStride, int K,
                                         const __bf16* __restrict__ wfrag,
                                         int colTiles, int colTile, int lane,
                                         v8f& acc0, v8f& acc1) {
    const int half = lane >> 4;
    const int row  = lane & 15;
    const int nk   = K >> 5;
    for (int kc = 0; kc < nk; ++kc) {
        v8u ai0, ai1;
#pragma unroll
        for (int v = 0; v < 8; ++v) {
            int kbase = kc * 32 + frag_kbase(half, v);   // even -> dword aligned
            ai0[v] = *(const unsigned int*)(a + row * aStride + kbase);
            ai1[v] = *(const unsigned int*)(a + (row + 16) * aStride + kbase);
        }
        v8u bi = *(const v8u*)(wfrag + (((size_t)kc * colTiles + colTile) * 32 + lane) * 16);
        bf16x16 bfrag = __builtin_bit_cast(bf16x16, bi);
        acc0 = __builtin_amdgcn_wmma_f32_16x16x32_bf16(false, __builtin_bit_cast(bf16x16, ai0),
                                                       false, bfrag, (short)0, acc0, false, false);
        acc1 = __builtin_amdgcn_wmma_f32_16x16x32_bf16(false, __builtin_bit_cast(bf16x16, ai1),
                                                       false, bfrag, (short)0, acc1, false, false);
    }
}

// Layer helpers: 32 rows x 128 cols, 8 waves (wave == colTile).
__device__ __forceinline__ void layer_relu32(const __bf16* a, int aStride, int K,
                                             const __bf16* wfrag, const float* bias,
                                             __bf16* outLDS, int wave, int lane) {
    v8f acc0 = {}, acc1 = {};
    mm_tile2(a, aStride, K, wfrag, 8, wave, lane, acc0, acc1);
    int col  = wave * 16 + (lane & 15);
    int half = lane >> 4;
    float b = bias[col];
#pragma unroll
    for (int r = 0; r < 8; ++r) {
        outLDS[(half * 8 + r) * 128 + col]        = (__bf16)fmaxf(acc0[r] + b, 0.0f);
        outLDS[(16 + half * 8 + r) * 128 + col]   = (__bf16)fmaxf(acc1[r] + b, 0.0f);
    }
}

__device__ __forceinline__ void layer_f32_32(const __bf16* a, int aStride, int K,
                                             const __bf16* wfrag, const float* bias,
                                             float* outF32, int wave, int lane) {
    v8f acc0 = {}, acc1 = {};
    mm_tile2(a, aStride, K, wfrag, 8, wave, lane, acc0, acc1);
    int col  = wave * 16 + (lane & 15);
    int half = lane >> 4;
    float b = bias[col];
#pragma unroll
    for (int r = 0; r < 8; ++r) {
        outF32[(half * 8 + r) * 128 + col]      = acc0[r] + b;
        outF32[(16 + half * 8 + r) * 128 + col] = acc1[r] + b;
    }
}

// LayerNorm stats over 128 cols for 32 rows (8 threads/row, keras eps=1e-3).
__device__ __forceinline__ void ln_stats32(const float* xf, float* mean, float* rsig,
                                           float* scratch) {
    int t = threadIdx.x, row = t >> 3, l = t & 7;
    float s = 0.f;
#pragma unroll
    for (int j = 0; j < 16; ++j) s += xf[row * 128 + l + j * 8];
    scratch[t] = s;
    __syncthreads();
    if (t < 32) {
        float m = 0.f;
#pragma unroll
        for (int j = 0; j < 8; ++j) m += scratch[t * 8 + j];
        mean[t] = m * (1.0f / 128.0f);
    }
    __syncthreads();
    float mu = mean[row];
    float v = 0.f;
#pragma unroll
    for (int j = 0; j < 16; ++j) {
        float d = xf[row * 128 + l + j * 8] - mu;
        v += d * d;
    }
    scratch[t] = v;
    __syncthreads();
    if (t < 32) {
        float vv = 0.f;
#pragma unroll
        for (int j = 0; j < 8; ++j) vv += scratch[t * 8 + j];
        rsig[t] = rsqrtf(vv * (1.0f / 128.0f) + 1e-3f);
    }
    __syncthreads();
}

// ---------------------------------------------------------------------------
// Encoder: feat[nRows][inDim] -> LN(MLP3) -> outF32 and/or outBf [nRows][128]
// ---------------------------------------------------------------------------
__global__ void k_encode(const float* __restrict__ feat, int inDim, int nRows,
                         const __bf16* __restrict__ w0, const __bf16* __restrict__ w1,
                         const __bf16* __restrict__ w2,
                         const float* __restrict__ b0, const float* __restrict__ b1,
                         const float* __restrict__ b2,
                         const float* __restrict__ g, const float* __restrict__ be,
                         float* __restrict__ outF32, __bf16* __restrict__ outBf) {
    __shared__ __bf16 aStage[32 * 32];
    __shared__ __bf16 h1[32 * 128];
    __shared__ __bf16 h2[32 * 128];
    __shared__ float  xf[32 * 128];
    __shared__ float  mean[32], rsig[32], scratch[256];

    int rowBase = blockIdx.x * 32;
    for (int f = threadIdx.x; f < 32 * 32; f += 256) {
        int row = f >> 5, c = f & 31;
        int rIdx = rowBase + row;
        float v = (rIdx < nRows && c < inDim) ? feat[(size_t)rIdx * inDim + c] : 0.0f;
        aStage[f] = (__bf16)v;
    }
    __syncthreads();
    int wave = threadIdx.x >> 5, lane = threadIdx.x & 31;
    layer_relu32(aStage, 32, 32, w0, b0, h1, wave, lane);
    __syncthreads();
    layer_relu32(h1, 128, 128, w1, b1, h2, wave, lane);
    __syncthreads();
    layer_f32_32(h2, 128, 128, w2, b2, xf, wave, lane);
    __syncthreads();
    ln_stats32(xf, mean, rsig, scratch);
    int row = threadIdx.x >> 3, l = threadIdx.x & 7;
    int rIdx = rowBase + row;
    if (rIdx < nRows) {
        float mu = mean[row], rs = rsig[row];
#pragma unroll
        for (int j = 0; j < 16; ++j) {
            int col = l + j * 8;
            float x = xf[row * 128 + col];
            float y = g[col] * (x - mu) * rs + be[col];
            if (outF32) outF32[(size_t)rIdx * 128 + col] = y;
            if (outBf)  outBf[(size_t)rIdx * 128 + col]  = (__bf16)y;
        }
    }
}

// ---------------------------------------------------------------------------
// Edge message: gather bf16 [h[rcv]|h[snd]|e] (K=384) -> LN(MLP3) -> atomic sum
// ---------------------------------------------------------------------------
__global__ void k_msg(const __bf16* __restrict__ h_bf, const __bf16* __restrict__ e_bf,
                      const int* __restrict__ snd, const int* __restrict__ rcv,
                      const __bf16* __restrict__ w0, const __bf16* __restrict__ w1,
                      const __bf16* __restrict__ w2,
                      const float* __restrict__ b0, const float* __restrict__ b1,
                      const float* __restrict__ b2,
                      const float* __restrict__ g, const float* __restrict__ be,
                      float* __restrict__ pooled, int E) {
    __shared__ __bf16 aStage[32 * 384];
    __shared__ __bf16 h1[32 * 128];
    __shared__ __bf16 h2[32 * 128];
    __shared__ float  xf[32 * 128];
    __shared__ float  mean[32], rsig[32], scratch[256];
    __shared__ int    ridx[32], sidx[32];

    int eBase = blockIdx.x * 32;
    if (threadIdx.x < 32) {
        int e = eBase + threadIdx.x;
        ridx[threadIdx.x] = (e < E) ? rcv[e] : 0;
        sidx[threadIdx.x] = (e < E) ? snd[e] : 0;
    }
    __syncthreads();
    // bf16 gather as dwords: rows are 64 dwords of h_bf / e_bf, 192 dwords staged
    const unsigned int* hb = (const unsigned int*)h_bf;
    const unsigned int* eb = (const unsigned int*)e_bf;
    unsigned int*       aU = (unsigned int*)aStage;
    for (int f = threadIdx.x; f < 32 * 192; f += 256) {
        int row = f / 192, c = f - row * 192;
        int e = eBase + row;
        unsigned int v = 0u;
        if (e < E) {
            if (c < 64)       v = hb[(size_t)ridx[row] * 64 + c];
            else if (c < 128) v = hb[(size_t)sidx[row] * 64 + (c - 64)];
            else              v = eb[(size_t)e * 64 + (c - 128)];
        }
        aU[f] = v;
    }
    __syncthreads();
    int wave = threadIdx.x >> 5, lane = threadIdx.x & 31;
    layer_relu32(aStage, 384, 384, w0, b0, h1, wave, lane);
    __syncthreads();
    layer_relu32(h1, 128, 128, w1, b1, h2, wave, lane);
    __syncthreads();
    layer_f32_32(h2, 128, 128, w2, b2, xf, wave, lane);
    __syncthreads();
    ln_stats32(xf, mean, rsig, scratch);
    int row = threadIdx.x >> 3, l = threadIdx.x & 7;
    int e = eBase + row;
    if (e < E) {
        int r = ridx[row];
        float mu = mean[row], rs = rsig[row];
#pragma unroll
        for (int j = 0; j < 16; ++j) {
            int col = l + j * 8;
            float x = xf[row * 128 + col];
            float y = g[col] * (x - mu) * rs + be[col];
            atomicAdd(&pooled[(size_t)r * 128 + col], y);
        }
    }
}

// ---------------------------------------------------------------------------
// Node update: [h(bf16) | pooled] (K=256) -> LN(MLP3) + residual(f32) -> hOut(+bf16)
// ---------------------------------------------------------------------------
__global__ void k_upd(const float* __restrict__ h, const __bf16* __restrict__ h_bf,
                      const float* __restrict__ pooled,
                      const __bf16* __restrict__ w0, const __bf16* __restrict__ w1,
                      const __bf16* __restrict__ w2,
                      const float* __restrict__ b0, const float* __restrict__ b1,
                      const float* __restrict__ b2,
                      const float* __restrict__ g, const float* __restrict__ be,
                      float* __restrict__ hOut, __bf16* __restrict__ hOutBf, int N) {
    __shared__ __bf16 aStage[32 * 256];
    __shared__ __bf16 h1[32 * 128];
    __shared__ __bf16 h2[32 * 128];
    __shared__ float  xf[32 * 128];
    __shared__ float  mean[32], rsig[32], scratch[256];

    int rowBase = blockIdx.x * 32;
    // h half: direct dword copy from bf16 mirror
    const unsigned int* hb = (const unsigned int*)h_bf;
    unsigned int*       aU = (unsigned int*)aStage;     // 128 dwords per row
    for (int f = threadIdx.x; f < 32 * 64; f += 256) {
        int row = f >> 6, c = f & 63;
        int rIdx = rowBase + row;
        aU[row * 128 + c] = (rIdx < N) ? hb[(size_t)rIdx * 64 + c] : 0u;
    }
    // pooled half: f32 -> bf16
    for (int f = threadIdx.x; f < 32 * 128; f += 256) {
        int row = f >> 7, c = f & 127;
        int rIdx = rowBase + row;
        float v = (rIdx < N) ? pooled[(size_t)rIdx * 128 + c] : 0.0f;
        aStage[row * 256 + 128 + c] = (__bf16)v;
    }
    __syncthreads();
    int wave = threadIdx.x >> 5, lane = threadIdx.x & 31;
    layer_relu32(aStage, 256, 256, w0, b0, h1, wave, lane);
    __syncthreads();
    layer_relu32(h1, 128, 128, w1, b1, h2, wave, lane);
    __syncthreads();
    layer_f32_32(h2, 128, 128, w2, b2, xf, wave, lane);
    __syncthreads();
    ln_stats32(xf, mean, rsig, scratch);
    int row = threadIdx.x >> 3, l = threadIdx.x & 7;
    int rIdx = rowBase + row;
    if (rIdx < N) {
        float mu = mean[row], rs = rsig[row];
#pragma unroll
        for (int j = 0; j < 16; ++j) {
            int col = l + j * 8;
            float x = xf[row * 128 + col];
            float y = h[(size_t)rIdx * 128 + col] + g[col] * (x - mu) * rs + be[col];
            hOut[(size_t)rIdx * 128 + col]   = y;
            hOutBf[(size_t)rIdx * 128 + col] = (__bf16)y;
        }
    }
}

// ---------------------------------------------------------------------------
// Decoder: MLP3 128->128->128->3 (last layer col-padded to 16, wave 0 only)
// ---------------------------------------------------------------------------
__global__ void k_dec(const __bf16* __restrict__ h_bf,
                      const __bf16* __restrict__ w0, const __bf16* __restrict__ w1,
                      const __bf16* __restrict__ w2,
                      const float* __restrict__ b0, const float* __restrict__ b1,
                      const float* __restrict__ b2,
                      float* __restrict__ out, int N) {
    __shared__ __bf16 aStage[32 * 128];
    __shared__ __bf16 h1[32 * 128];
    __shared__ __bf16 h2[32 * 128];

    int rowBase = blockIdx.x * 32;
    const unsigned int* hb = (const unsigned int*)h_bf;
    unsigned int*       aU = (unsigned int*)aStage;
    for (int f = threadIdx.x; f < 32 * 64; f += 256) {
        int row = f >> 6, c = f & 63;
        int rIdx = rowBase + row;
        aU[f] = (rIdx < N) ? hb[(size_t)rIdx * 64 + c] : 0u;
    }
    __syncthreads();
    int wave = threadIdx.x >> 5, lane = threadIdx.x & 31;
    layer_relu32(aStage, 128, 128, w0, b0, h1, wave, lane);
    __syncthreads();
    layer_relu32(h1, 128, 128, w1, b1, h2, wave, lane);
    __syncthreads();
    if (wave == 0) {   // wave-uniform branch: EXEC all ones inside
        v8f acc0 = {}, acc1 = {};
        mm_tile2(h2, 128, 128, w2, 1, 0, lane, acc0, acc1);
        int col = lane & 15, half = lane >> 4;
        if (col < 3) {
            float b = b2[col];
#pragma unroll
            for (int r = 0; r < 8; ++r) {
                int n0 = rowBase + half * 8 + r;
                int n1 = rowBase + 16 + half * 8 + r;
                if (n0 < N) out[(size_t)n0 * 3 + col] = acc0[r] + b;
                if (n1 < N) out[(size_t)n1 * 3 + col] = acc1[r] + b;
            }
        }
    }
}

// ---------------------------------------------------------------------------
// Host orchestration
// ---------------------------------------------------------------------------
extern "C" void kernel_launch(void* const* d_in, const int* in_sizes, int n_in,
                              void* d_out, int out_size, void* d_ws, size_t ws_size,
                              hipStream_t stream) {
    const int NODE_IN = 30, EDGE_IN = 3, D = 128;
    const float* node_feat = (const float*)d_in[0];
    const float* edge_feat = (const float*)d_in[1];
    const int*   senders   = (const int*)d_in[2];
    const int*   receivers = (const int*)d_in[3];
    const int N = in_sizes[0] / NODE_IN;
    const int E = in_sizes[2];
    int T = in_sizes[21] / D;          // msg_b0 is [T,128]
    if (T > 16) T = 16;

    const float* encn_W0 = (const float*)d_in[4];
    const float* encn_b0 = (const float*)d_in[5];
    const float* encn_W1 = (const float*)d_in[6];
    const float* encn_b1 = (const float*)d_in[7];
    const float* encn_W2 = (const float*)d_in[8];
    const float* encn_b2 = (const float*)d_in[9];
    const float* encn_g  = (const float*)d_in[10];
    const float* encn_be = (const float*)d_in[11];
    const float* ence_W0 = (const float*)d_in[12];
    const float* ence_b0 = (const float*)d_in[13];
    const float* ence_W1 = (const float*)d_in[14];
    const float* ence_b1 = (const float*)d_in[15];
    const float* ence_W2 = (const float*)d_in[16];
    const float* ence_b2 = (const float*)d_in[17];
    const float* ence_g  = (const float*)d_in[18];
    const float* ence_be = (const float*)d_in[19];
    const float* msg_W0  = (const float*)d_in[20];
    const float* msg_b0  = (const float*)d_in[21];
    const float* msg_W1  = (const float*)d_in[22];
    const float* msg_b1  = (const float*)d_in[23];
    const float* msg_W2  = (const float*)d_in[24];
    const float* msg_b2  = (const float*)d_in[25];
    const float* msg_g   = (const float*)d_in[26];
    const float* msg_be  = (const float*)d_in[27];
    const float* upd_W0  = (const float*)d_in[28];
    const float* upd_b0  = (const float*)d_in[29];
    const float* upd_W1  = (const float*)d_in[30];
    const float* upd_b1  = (const float*)d_in[31];
    const float* upd_W2  = (const float*)d_in[32];
    const float* upd_b2  = (const float*)d_in[33];
    const float* upd_g   = (const float*)d_in[34];
    const float* upd_be  = (const float*)d_in[35];
    const float* dec_W0  = (const float*)d_in[36];
    const float* dec_b0  = (const float*)d_in[37];
    const float* dec_W1  = (const float*)d_in[38];
    const float* dec_b1  = (const float*)d_in[39];
    const float* dec_W2  = (const float*)d_in[40];
    const float* dec_b2  = (const float*)d_in[41];

    // ---- workspace carve-up ----
    char* ws = (char*)d_ws;
    size_t off = 0;
    auto alloc = [&](size_t bytes) -> void* {
        off = (off + 63) & ~(size_t)63;
        void* p = ws + off;
        off += bytes;
        return p;
    };
    float*  h_a    = (float*)alloc((size_t)N * D * sizeof(float));
    float*  h_b    = (float*)alloc((size_t)N * D * sizeof(float));
    __bf16* hb_a   = (__bf16*)alloc((size_t)N * D * sizeof(__bf16));
    __bf16* hb_b   = (__bf16*)alloc((size_t)N * D * sizeof(__bf16));
    __bf16* e_bf   = (__bf16*)alloc((size_t)E * D * sizeof(__bf16));
    float*  pooled = (float*)alloc((size_t)N * D * sizeof(float));
    auto allocW = [&](int Kpad, int Npad) -> __bf16* {
        return (__bf16*)alloc((size_t)(Kpad / 32) * (Npad / 16) * 512 * sizeof(__bf16));
    };
    auto conv = [&](const float* W, __bf16* dst, int K, int Kpad, int Nc, int Npad) {
        int total = (Kpad / 32) * (Npad / 16) * 512;
        k_convert_w<<<(total + 255) / 256, 256, 0, stream>>>(W, dst, K, Kpad, Nc, Npad);
    };

    __bf16* wencn0 = allocW(32, 128);  conv(encn_W0, wencn0, NODE_IN, 32, 128, 128);
    __bf16* wencn1 = allocW(128, 128); conv(encn_W1, wencn1, 128, 128, 128, 128);
    __bf16* wencn2 = allocW(128, 128); conv(encn_W2, wencn2, 128, 128, 128, 128);
    __bf16* wence0 = allocW(32, 128);  conv(ence_W0, wence0, EDGE_IN, 32, 128, 128);
    __bf16* wence1 = allocW(128, 128); conv(ence_W1, wence1, 128, 128, 128, 128);
    __bf16* wence2 = allocW(128, 128); conv(ence_W2, wence2, 128, 128, 128, 128);

    __bf16 *wm0[16], *wm1[16], *wm2[16], *wu0[16], *wu1[16], *wu2[16];
    for (int t = 0; t < T; ++t) {
        wm0[t] = allocW(384, 128); conv(msg_W0 + (size_t)t * 384 * 128, wm0[t], 384, 384, 128, 128);
        wm1[t] = allocW(128, 128); conv(msg_W1 + (size_t)t * 128 * 128, wm1[t], 128, 128, 128, 128);
        wm2[t] = allocW(128, 128); conv(msg_W2 + (size_t)t * 128 * 128, wm2[t], 128, 128, 128, 128);
        wu0[t] = allocW(256, 128); conv(upd_W0 + (size_t)t * 256 * 128, wu0[t], 256, 256, 128, 128);
        wu1[t] = allocW(128, 128); conv(upd_W1 + (size_t)t * 128 * 128, wu1[t], 128, 128, 128, 128);
        wu2[t] = allocW(128, 128); conv(upd_W2 + (size_t)t * 128 * 128, wu2[t], 128, 128, 128, 128);
    }
    __bf16* wdec0 = allocW(128, 128); conv(dec_W0, wdec0, 128, 128, 128, 128);
    __bf16* wdec1 = allocW(128, 128); conv(dec_W1, wdec1, 128, 128, 128, 128);
    __bf16* wdec2 = allocW(128, 16);  conv(dec_W2, wdec2, 128, 128, 3, 16);

    const int nodeBlocks = (N + 31) / 32;
    const int edgeBlocks = (E + 31) / 32;

    // Encoders: node -> f32 master + bf16 mirror; edge -> bf16 only
    k_encode<<<nodeBlocks, 256, 0, stream>>>(node_feat, NODE_IN, N,
        wencn0, wencn1, wencn2, encn_b0, encn_b1, encn_b2, encn_g, encn_be,
        h_a, hb_a);
    k_encode<<<edgeBlocks, 256, 0, stream>>>(edge_feat, EDGE_IN, E,
        wence0, wence1, wence2, ence_b0, ence_b1, ence_b2, ence_g, ence_be,
        (float*)nullptr, e_bf);

    // Processor rounds
    float*  h_cur  = h_a;  float*  h_nxt  = h_b;
    __bf16* hb_cur = hb_a; __bf16* hb_nxt = hb_b;
    long pooledN = (long)N * D;
    for (int t = 0; t < T; ++t) {
        k_zero<<<(int)((pooledN + 255) / 256), 256, 0, stream>>>(pooled, pooledN);
        k_msg<<<edgeBlocks, 256, 0, stream>>>(hb_cur, e_bf, senders, receivers,
            wm0[t], wm1[t], wm2[t],
            msg_b0 + (size_t)t * D, msg_b1 + (size_t)t * D, msg_b2 + (size_t)t * D,
            msg_g + (size_t)t * D, msg_be + (size_t)t * D, pooled, E);
        k_upd<<<nodeBlocks, 256, 0, stream>>>(h_cur, hb_cur, pooled,
            wu0[t], wu1[t], wu2[t],
            upd_b0 + (size_t)t * D, upd_b1 + (size_t)t * D, upd_b2 + (size_t)t * D,
            upd_g + (size_t)t * D, upd_be + (size_t)t * D, h_nxt, hb_nxt, N);
        float*  tf = h_cur;  h_cur  = h_nxt;  h_nxt  = tf;
        __bf16* tb = hb_cur; hb_cur = hb_nxt; hb_nxt = tb;
    }

    // Decoder
    k_dec<<<nodeBlocks, 256, 0, stream>>>(hb_cur, wdec0, wdec1, wdec2,
        dec_b0, dec_b1, dec_b2, (float*)d_out, N);
}
